// MoDL_CG_11639361372535
// MI455X (gfx1250) — matrix operand order repaired and checked by reading, not compiled, not run on hardware
//
#include <hip/hip_runtime.h>
#include <hip/hip_bf16.h>

// ---------------- types ----------------
typedef __attribute__((ext_vector_type(16))) _Float16 v16h;
typedef __attribute__((ext_vector_type(8)))  float    v8f;
typedef __attribute__((ext_vector_type(2)))  float    v2f;
typedef int v4i __attribute__((vector_size(16)));      // matches async-LDS builtin param

union U16x8 { uint4 u; _Float16 h[8]; };

static constexpr int cH = 512;
static constexpr int cW = 384;
static constexpr int cC = 16;
static constexpr int cNK = 16;
static constexpr int TILES_X = cW / 16;             // 24
static constexpr int NTILES  = cH * TILES_X;        // 12288 wave-tiles
static constexpr int CONV_BLOCKS = NTILES / 8;      // 1536 (8 waves / block, 3 blocks/row)
static constexpr int GEMM_TILES  = (cW/16)*(cW/16); // 576
static constexpr int GEMM_BLOCKS = GEMM_TILES / 8;  // 72
static constexpr int ALPHA_BLOCKS = 144;
static constexpr int HALO_W = 130;                  // 128 pixels + 1 halo each side

// ---------------- ws layout (bytes) ----------------
enum : size_t {
  OFF_SS    = 0,                 // float  [H*W]            786432
  OFF_AHB   = 786432,            // float2 [H*W]           1572864
  OFF_IMG   = 2359296,           // float2 [H*W]
  OFF_D     = 3932160,           // float2 [H*W]
  OFF_QD    = 5505024,           // float2 [H*W]
  OFF_P     = 7077888,           // float2 [384*384]       1179648
  OFF_Q     = 8257536,           // float2 [384*384]
  OFF_SMALL = 9437184,           // 8 KB small scalars
  OFF_WTS   = 9445376,           // 16 KB packed f16 weights
  OFF_Y0    = 9461760,           // f16 [H][W][16]         6291456
  OFF_Y1    = 15753216,
  OFF_Y2    = 22044672,
  OFF_YCOIL = 9461760,           // float2 [C*H*W] aliases Y0..Y2 (dead after decode)
  OFF_BNSUM = 34627584,          // float [CONV_BLOCKS*16]
  OFF_BNSQ  = 34725888,          // float [CONV_BLOCKS*16]  -> end ~34.9 MB
};
// small region floats: s1@0 t1@16 s2@32 t2@48 alpha@64 alphaPart@128..415

// ---------------- helpers ----------------
__device__ __forceinline__ float2 c2(float x, float y){ float2 r; r.x=x; r.y=y; return r; }
__device__ __forceinline__ float2 cadd(float2 a, float2 b){ return c2(a.x+b.x, a.y+b.y); }
__device__ __forceinline__ float2 cmul(float2 a, float2 b){ return c2(a.x*b.x-a.y*b.y, a.x*b.y+a.y*b.x); }

__device__ __forceinline__ v8f wmma_f16(v16h a, v16h b, v8f c) {
  return __builtin_amdgcn_wmma_f32_16x16x32_f16(false, a, false, b, (short)0, c, false, false);
}
#if __has_builtin(__builtin_amdgcn_wmma_f32_16x16x4_f32)
__device__ __forceinline__ v8f wmma_f32(v2f a, v2f b, v8f c) {
  return __builtin_amdgcn_wmma_f32_16x16x4_f32(false, a, false, b, (short)0, c, false, false);
}
#else
__device__ __forceinline__ v8f wmma_f32(v2f a, v2f b, v8f c) {
  #pragma unroll
  for (int r = 0; r < 8; ++r) c[r] = fmaf(a[0], b[0], fmaf(a[1], b[1], c[r]));
  return c;
}
#endif

// -------- async global->LDS staging (CDNA5 ASYNCcnt path, guarded) --------
#if __has_builtin(__builtin_amdgcn_global_load_async_to_lds_b128)
#define HAVE_ASYNC_LDS 1
#else
#define HAVE_ASYNC_LDS 0
#endif

__device__ __forceinline__ void wait_async_lds() {
#if HAVE_ASYNC_LDS
#if __has_builtin(__builtin_amdgcn_s_wait_asynccnt)
  __builtin_amdgcn_s_wait_asynccnt(0);
#else
  asm volatile("s_wait_asynccnt 0x0" ::: "memory");
#endif
#endif
}

// Stage rows ty-1..ty+1, cols xs-1..xs+128, all 16 channels, into LDS.
// 3*130 pixels * 2 16-byte chunks = 780 chunks, cooperatively by 256 threads.
__device__ __forceinline__ void stage_tile(const _Float16* __restrict__ yin,
                                           _Float16* lds, int ty, int xs) {
  for (int e = threadIdx.x; e < 3 * HALO_W * 2; e += 256) {
    int pix = e >> 1, c0 = (e & 1) * 8;
    int pr = pix / HALO_W, pc = pix - pr * HALO_W;
    int gy = ty - 1 + pr, gx = xs - 1 + pc;
    int loff = (pr * HALO_W + pc) * cNK + c0;          // in halves
    if (gy >= 0 && gy < cH && gx >= 0 && gx < cW) {
      const _Float16* src = yin + ((size_t)gy * cW + gx) * cNK + c0;
#if HAVE_ASYNC_LDS
      __builtin_amdgcn_global_load_async_to_lds_b128(
          (__attribute__((address_space(1))) v4i*)(src),
          (__attribute__((address_space(3))) v4i*)(lds + loff), 0, 0);
#else
      *(uint4*)(lds + loff) = *(const uint4*)src;
#endif
    } else {
      uint4 z = {0u, 0u, 0u, 0u};
      *(uint4*)(lds + loff) = z;
    }
  }
  wait_async_lds();
  __syncthreads();
}

// ---------------- weight packing ----------------
// B layout for v_wmma_*_16x16x32: lane L holds column N=L%16 of the 32x16 slab,
// 16 contiguous halves = K in table order (h=L/16).  K map: K = tap*Cin + cin (Cin=16 mid, 2 first).
__global__ void k_prep_w(const float* __restrict__ w0, const float* __restrict__ w1,
                         const float* __restrict__ w2, const float* __restrict__ wl,
                         _Float16* __restrict__ B0, _Float16* __restrict__ B1,
                         _Float16* __restrict__ B2, _Float16* __restrict__ BL) {
  int t = threadIdx.x;
  for (int e = t; e < 32*16; e += 256) {
    int L = e >> 4, j = e & 15, n = L & 15, K = ((j & 8) << 1) + (L >> 4) * 8 + (j & 7);
    float v = 0.f;
    if (K < 18) { int tap = K >> 1, cin = K & 1; v = w0[((n*2+cin)*3 + tap/3)*3 + tap%3]; }
    B0[e] = (_Float16)v;
  }
  for (int e = t; e < 160*16; e += 256) {
    int L = (e >> 4) & 31, j = e & 15, n = L & 15, K = (e >> 9) * 32 + ((j & 8) << 1) + (L >> 4) * 8 + (j & 7);
    float v1 = 0.f, v2 = 0.f, vL = 0.f;
    if (K < 144) {
      int tap = K >> 4, cin = K & 15;
      int wi = ((n*16+cin)*3 + tap/3)*3 + tap%3;
      v1 = w1[wi]; v2 = w2[wi];
      if (n < 2) vL = wl[wi];
    }
    B1[e] = (_Float16)v1; B2[e] = (_Float16)v2; BL[e] = (_Float16)vL;
  }
}

// ---------------- IFFT (ortho) : Stockham in LDS ----------------
__global__ void __launch_bounds__(256) k_fft_col512(const float2* __restrict__ src,
                                                    float2* __restrict__ dst, float dir) {
  __shared__ float2 buf0[512], buf1[512];
  int cc = blockIdx.x / cW, w = blockIdx.x % cW;
  size_t base = (size_t)cc * cH * cW + w;
  int tid = threadIdx.x;
  buf0[tid]       = src[base + (size_t)tid * cW];
  buf0[tid + 256] = src[base + (size_t)(tid + 256) * cW];
  __syncthreads();
  float2* cur = buf0; float2* nxt = buf1;
  int n = 512, s = 1;
  while (n > 1) {
    int nh = n >> 1;
    int p = tid / s, q = tid - p * s;
    float2 A = cur[q + s*p];
    float2 Bv = cur[q + s*(p + nh)];
    float ang = dir * 6.28318530717958647f * (float)p / (float)n;
    float sn, cs; __sincosf(ang, &sn, &cs);
    float2 dif = c2(A.x - Bv.x, A.y - Bv.y);
    nxt[q + s*2*p]     = c2(A.x + Bv.x, A.y + Bv.y);
    nxt[q + s*(2*p+1)] = c2(dif.x*cs - dif.y*sn, dif.x*sn + dif.y*cs);
    __syncthreads();
    float2* tb = cur; cur = nxt; nxt = tb;
    n = nh; s <<= 1;
  }
  float scl = rsqrtf(512.f);
  float2 r0 = cur[tid], r1 = cur[tid + 256];
  dst[base + (size_t)tid * cW]        = c2(r0.x*scl, r0.y*scl);
  dst[base + (size_t)(tid + 256)*cW]  = c2(r1.x*scl, r1.y*scl);
}

__global__ void __launch_bounds__(192) k_fft_row384(float2* __restrict__ data, float dir) {
  __shared__ float2 buf0[384], buf1[384];
  size_t base = (size_t)blockIdx.x * cW;
  int tid = threadIdx.x;
  buf0[tid] = data[base + tid]; buf0[tid + 192] = data[base + tid + 192];
  __syncthreads();
  float2* cur = buf0; float2* nxt = buf1;
  int n = 384, s = 1;
  while (n > 1) {
    if (n % 3 == 0) {
      int nt = n / 3, cnt = nt * s;
      const float s3 = 0.86602540378443865f;
      float2 c31 = c2(-0.5f,  dir * s3);
      float2 c32 = c2(-0.5f, -dir * s3);
      for (int t = tid; t < cnt; t += 192) {
        int p = t / s, q = t - p * s;
        float2 t0 = cur[q + s*p], t1 = cur[q + s*(p+nt)], t2 = cur[q + s*(p+2*nt)];
        float2 u0 = cadd(t0, cadd(t1, t2));
        float2 u1 = cadd(t0, cadd(cmul(c31, t1), cmul(c32, t2)));
        float2 u2 = cadd(t0, cadd(cmul(c32, t1), cmul(c31, t2)));
        float ang = dir * 6.28318530717958647f * (float)p / (float)n;
        float sn, cs; __sincosf(ang, &sn, &cs);
        float2 w1 = c2(cs, sn);
        float2 w2 = cmul(w1, w1);
        nxt[q + s*3*p]     = u0;
        nxt[q + s*(3*p+1)] = cmul(u1, w1);
        nxt[q + s*(3*p+2)] = cmul(u2, w2);
      }
      n = nt; s *= 3;
    } else {
      int nh = n >> 1, cnt = nh * s;
      for (int t = tid; t < cnt; t += 192) {
        int p = t / s, q = t - p * s;
        float2 A = cur[q + s*p], Bv = cur[q + s*(p+nh)];
        float ang = dir * 6.28318530717958647f * (float)p / (float)n;
        float sn, cs; __sincosf(ang, &sn, &cs);
        float2 dif = c2(A.x - Bv.x, A.y - Bv.y);
        nxt[q + s*2*p]     = c2(A.x + Bv.x, A.y + Bv.y);
        nxt[q + s*(2*p+1)] = c2(dif.x*cs - dif.y*sn, dif.x*sn + dif.y*cs);
      }
      n = nh; s <<= 1;
    }
    __syncthreads();
    float2* tb = cur; cur = nxt; nxt = tb;
  }
  float scl = rsqrtf(384.f);
  float2 r0 = cur[tid], r1 = cur[tid + 192];
  data[base + tid]       = c2(r0.x*scl, r0.y*scl);
  data[base + tid + 192] = c2(r1.x*scl, r1.y*scl);
}

// ---------------- coil combine: Ahb, image, SS = sum|S|^2 ----------------
__global__ void k_decode(const float2* __restrict__ yc, const float* __restrict__ smaps,
                         float2* __restrict__ Ahb, float2* __restrict__ img,
                         float* __restrict__ SSm) {
  int p = blockIdx.x * blockDim.x + threadIdx.x;
  if (p >= cH * cW) return;
  float2 acc = c2(0.f, 0.f); float ss = 0.f;
  for (int cc = 0; cc < cC; ++cc) {
    const float* sp = smaps + ((size_t)cc * cH * cW + p) * 2;
    float sr = sp[0], si = sp[1];
    float2 v = yc[(size_t)cc * cH * cW + p];
    acc.x += sr * v.x + si * v.y;        // conj(S)*v
    acc.y += sr * v.y - si * v.x;
    ss += sr * sr + si * si;
  }
  Ahb[p] = acc; img[p] = acc; SSm[p] = ss;
}

// ---------------- conv layer 0 : 2ch -> 16ch, one WMMA slab ----------------
__global__ void __launch_bounds__(256) k_conv0(const float2* __restrict__ img,
                                               const _Float16* __restrict__ B0,
                                               const float* __restrict__ b0,
                                               _Float16* __restrict__ y0) {
  int wid  = (blockIdx.x * blockDim.x + threadIdx.x) >> 5;
  int lane = threadIdx.x & 31;
  int ty = wid / TILES_X, x0 = (wid % TILES_X) * 16;
  int m = lane & 15, kh = lane >> 4;
  int x = x0 + m;
  v16h a;
  #pragma unroll
  for (int j = 0; j < 16; ++j) {
    int K = ((j & 8) << 1) + kh * 8 + (j & 7);
    float v = 0.f;
    if (K < 18) {
      int tap = K >> 1, cin = K & 1;
      int yy = ty + tap / 3 - 1, xx = x + tap % 3 - 1;
      if (yy >= 0 && yy < cH && xx >= 0 && xx < cW) {
        float2 pv = img[yy * cW + xx];
        v = cin ? pv.y : pv.x;
      }
    }
    a[j] = (_Float16)v;
  }
  v16h b = *(const v16h*)(B0 + lane * 16);
  v8f c = {};
  c = wmma_f16(a, b, c);
  int n = lane & 15;
  float bias = b0[n];
  #pragma unroll
  for (int r = 0; r < 8; ++r) {
    int mo = r + kh * 8;
    float v = fmaxf(c[r] + bias, 0.f);
    y0[((size_t)ty * cW + x0 + mo) * cNK + n] = (_Float16)v;
  }
}

// ---- A-fragment from staged LDS tile (no bounds checks; halo pre-zeroed) ----
__device__ __forceinline__ v16h build_a(const _Float16* lds, int lx, int c0,
                                        int s, bool act, const float* sv, const float* tv) {
  v16h a;
  #pragma unroll
  for (int hh = 0; hh < 2; ++hh) {
    int tap = 2 * s + hh;
    uint4 raw = {0u, 0u, 0u, 0u};
    if (tap < 9) {
      int pr = tap / 3;                 // dy + 1
      int pc = lx + tap % 3 - 1;        // halo-local column
      raw = *(const uint4*)(lds + (pr * HALO_W + pc) * cNK + c0);
    }
    U16x8 u; u.u = raw;
    if (act) {
      #pragma unroll
      for (int k2 = 0; k2 < 8; ++k2) {
        float v = (float)u.h[k2];
        u.h[k2] = (_Float16)fmaxf(fmaf(v, sv[k2], tv[k2]), 0.f);
      }
    }
    ((uint4*)&a)[hh] = u.u;
  }
  return a;
}

// ---------------- mid conv: 16->16, five WMMA slabs, LDS-staged input,
// per-block BN partials (deterministic) ----------------
__global__ void __launch_bounds__(256) k_conv_mid(const _Float16* __restrict__ yin,
                                                  const _Float16* __restrict__ Bm,
                                                  const float* __restrict__ bias,
                                                  const float* __restrict__ sIn,
                                                  const float* __restrict__ tIn,
                                                  _Float16* __restrict__ yout,
                                                  float* __restrict__ bnSum,
                                                  float* __restrict__ bnSq) {
  __shared__ __align__(16) _Float16 tile[3 * HALO_W * cNK];
  __shared__ float psum[256], psq[256];
  int ty = blockIdx.x / 3;
  int xs = (blockIdx.x % 3) * 128;
  stage_tile(yin, tile, ty, xs);

  int lane = threadIdx.x & 31;
  int wloc = threadIdx.x >> 5;
  int x0 = xs + wloc * 16;
  int m = lane & 15, kh = lane >> 4;
  int lx = wloc * 16 + m + 1;           // halo-local column of this pixel
  int c0 = kh * 8;
  bool act = (sIn != nullptr);
  float sv[8], tv[8];
  if (act) {
    #pragma unroll
    for (int k2 = 0; k2 < 8; ++k2) { sv[k2] = sIn[c0 + k2]; tv[k2] = tIn[c0 + k2]; }
  }
  v8f c = {};
  #pragma unroll
  for (int s = 0; s < 5; ++s) {
    v16h a = build_a(tile, lx, c0, s, act, sv, tv);
    v16h b = *(const v16h*)(Bm + (s * 32 + lane) * 16);
    c = wmma_f16(a, b, c);
  }
  int n = lane & 15;
  float bb = bias[n];
  float lsum = 0.f, lsq = 0.f;
  #pragma unroll
  for (int r = 0; r < 8; ++r) {
    int mo = r + kh * 8;
    float v = c[r] + bb;
    yout[((size_t)ty * cW + x0 + mo) * cNK + n] = (_Float16)v;
    lsum += v; lsq += v * v;
  }
  psum[threadIdx.x] = lsum; psq[threadIdx.x] = lsq;
  __syncthreads();
  if (threadIdx.x < 16) {         // deterministic fixed-order block reduce
    float a2 = 0.f, b2 = 0.f;
    #pragma unroll
    for (int w8 = 0; w8 < 8; ++w8) {
      int t0 = w8 * 32 + threadIdx.x;
      a2 += psum[t0] + psum[t0 + 16];
      b2 += psq[t0] + psq[t0 + 16];
    }
    bnSum[blockIdx.x * 16 + threadIdx.x] = a2;
    bnSq [blockIdx.x * 16 + threadIdx.x] = b2;
  }
}

// ---------------- BN params from partials ----------------
__global__ void k_bnp(const float* __restrict__ bnSum, const float* __restrict__ bnSq,
                      const float* __restrict__ g, const float* __restrict__ be,
                      float* __restrict__ sO, float* __restrict__ tO) {
  int c = threadIdx.x; if (c >= 16) return;
  float s = 0.f, q = 0.f;
  for (int bI = 0; bI < CONV_BLOCKS; ++bI) { s += bnSum[bI*16 + c]; q += bnSq[bI*16 + c]; }
  const float inv = 1.0f / (float)(cH * cW);
  float m = s * inv;
  float var = q * inv - m * m;
  float sc = g[c] * rsqrtf(var + 1e-5f);
  sO[c] = sc; tO[c] = be[c] - m * sc;
}

// ---------------- last conv: 16->2, fused residual subtract ----------------
__global__ void __launch_bounds__(256) k_convL(const _Float16* __restrict__ yin,
                                               const _Float16* __restrict__ BL,
                                               const float* __restrict__ sIn,
                                               const float* __restrict__ tIn,
                                               const float* __restrict__ bl,
                                               float2* __restrict__ img) {
  __shared__ __align__(16) _Float16 tile[3 * HALO_W * cNK];
  int ty = blockIdx.x / 3;
  int xs = (blockIdx.x % 3) * 128;
  stage_tile(yin, tile, ty, xs);

  int lane = threadIdx.x & 31;
  int wloc = threadIdx.x >> 5;
  int x0 = xs + wloc * 16;
  int m = lane & 15, kh = lane >> 4;
  int lx = wloc * 16 + m + 1;
  int c0 = kh * 8;
  float sv[8], tv[8];
  #pragma unroll
  for (int k2 = 0; k2 < 8; ++k2) { sv[k2] = sIn[c0 + k2]; tv[k2] = tIn[c0 + k2]; }
  v8f c = {};
  #pragma unroll
  for (int s = 0; s < 5; ++s) {
    v16h a = build_a(tile, lx, c0, s, true, sv, tv);
    v16h b = *(const v16h*)(BL + (s * 32 + lane) * 16);
    c = wmma_f16(a, b, c);
  }
  int n = lane & 15;
  if (n < 2) {
    float bb = bl[n];
    float* f = (float*)img;
    #pragma unroll
    for (int r = 0; r < 8; ++r) {
      int mo = r + kh * 8;
      int p = ty * cW + x0 + mo;
      f[p * 2 + n] = f[p * 2 + n] - (c[r] + bb);   // image = x - res
    }
  }
}

// ---------------- d = Ahb - (SS+lam) image ; Qd = (SS+lam) d ----------------
__global__ void k_make_dqd(const float2* __restrict__ img, const float2* __restrict__ Ahb,
                           const float* __restrict__ SSm, const float* __restrict__ lamA,
                           int it, float2* __restrict__ dv, float2* __restrict__ qv) {
  int p = blockIdx.x * blockDim.x + threadIdx.x;
  if (p >= cH * cW) return;
  float a = SSm[p] + lamA[it];
  float2 x = img[p], hb = Ahb[p];
  float2 d = c2(hb.x - a * x.x, hb.y - a * x.y);
  dv[p] = d;
  qv[p] = c2(a * d.x, a * d.y);
}

// ---------------- P = d^H d, Q = d^H Qd  (fp32 WMMA, complex via 4 real) ----
__global__ void __launch_bounds__(256) k_gemm_pq(const float2* __restrict__ dv,
                                                 const float2* __restrict__ qv,
                                                 float2* __restrict__ P,
                                                 float2* __restrict__ Q) {
  int wid  = (blockIdx.x * blockDim.x + threadIdx.x) >> 5;
  int lane = threadIdx.x & 31;
  int i0 = (wid / TILES_X) * 16, j0 = (wid % TILES_X) * 16;
  int li = lane & 15, kh = lane >> 4;
  int ia = i0 + li, jb = j0 + li;
  v8f Pre = {}, Pim = {}, Qre = {}, Qim = {};
  #pragma unroll 4
  for (int kb = 0; kb < cH; kb += 4) {
    int k0 = kb + 2 * kh;
    if (kb + 8 < cH) {
      __builtin_prefetch((const void*)(dv + (size_t)(k0 + 8) * cW + ia), 0, 1);
      __builtin_prefetch((const void*)(qv + (size_t)(k0 + 8) * cW + jb), 0, 1);
    }
    float2 a0 = dv[(size_t)k0 * cW + ia];
    float2 a1 = dv[(size_t)(k0 + 1) * cW + ia];
    v2f Ar  = { a0.x,  a1.x};
    v2f Ai  = { a0.y,  a1.y};
    v2f Ain = {-a0.y, -a1.y};
    float2 bd0 = dv[(size_t)k0 * cW + jb];
    float2 bd1 = dv[(size_t)(k0 + 1) * cW + jb];
    float2 bq0 = qv[(size_t)k0 * cW + jb];
    float2 bq1 = qv[(size_t)(k0 + 1) * cW + jb];
    v2f Brd = {bd0.x, bd1.x}, Bid = {bd0.y, bd1.y};
    v2f Brq = {bq0.x, bq1.x}, Biq = {bq0.y, bq1.y};
    // conj(a)*b : Re = ar*br + ai*bi ; Im = ar*bi - ai*br
    Pre = wmma_f32(Ar,  Brd, Pre);  Pre = wmma_f32(Ai,  Bid, Pre);
    Pim = wmma_f32(Ar,  Bid, Pim);  Pim = wmma_f32(Ain, Brd, Pim);
    Qre = wmma_f32(Ar,  Brq, Qre);  Qre = wmma_f32(Ai,  Biq, Qre);
    Qim = wmma_f32(Ar,  Biq, Qim);  Qim = wmma_f32(Ain, Brq, Qim);
  }
  #pragma unroll
  for (int r = 0; r < 8; ++r) {
    int ii = i0 + r + 8 * kh, jj = j0 + li;
    P[ii * cW + jj] = c2(Pre[r], Pim[r]);
    Q[ii * cW + jj] = c2(Qre[r], Qim[r]);
  }
}

// ---------------- alpha = sum(P/Q) : two-stage deterministic reduce --------
__global__ void __launch_bounds__(256) k_alpha(const float2* __restrict__ P,
                                               const float2* __restrict__ Q,
                                               float* __restrict__ part) {
  __shared__ float sre[256], sim[256];
  int idx = blockIdx.x * blockDim.x + threadIdx.x;
  float ar = 0.f, ai = 0.f;
  for (int p = idx; p < cW * cW; p += gridDim.x * blockDim.x) {
    float2 a = P[p], b = Q[p];
    float inv = 1.0f / (b.x * b.x + b.y * b.y);
    ar += (a.x * b.x + a.y * b.y) * inv;
    ai += (a.y * b.x - a.x * b.y) * inv;
  }
  sre[threadIdx.x] = ar; sim[threadIdx.x] = ai;
  __syncthreads();
  for (int st = 128; st > 0; st >>= 1) {
    if (threadIdx.x < st) { sre[threadIdx.x] += sre[threadIdx.x + st];
                            sim[threadIdx.x] += sim[threadIdx.x + st]; }
    __syncthreads();
  }
  if (threadIdx.x == 0) { part[blockIdx.x * 2] = sre[0]; part[blockIdx.x * 2 + 1] = sim[0]; }
}

__global__ void k_alpha_fin(const float* __restrict__ part, float2* __restrict__ alpha) {
  if (threadIdx.x == 0) {
    float a = 0.f, b = 0.f;
    for (int i = 0; i < ALPHA_BLOCKS; ++i) { a += part[2*i]; b += part[2*i + 1]; }
    *alpha = c2(a, b);
  }
}

// ---------------- image += alpha * d ----------------
__global__ void k_update(float2* __restrict__ img, const float2* __restrict__ dv,
                         const float2* __restrict__ alpha) {
  int p = blockIdx.x * blockDim.x + threadIdx.x;
  if (p >= cH * cW) return;
  float2 al = *alpha;
  float2 d = dv[p];
  img[p].x += al.x * d.x - al.y * d.y;
  img[p].y += al.x * d.y + al.y * d.x;
}

// ---------------- crop rows 64..447 -> out (1,384,384,2) ----------------
__global__ void k_crop(const float2* __restrict__ img, float* __restrict__ out) {
  int p = blockIdx.x * blockDim.x + threadIdx.x;
  if (p >= cW * cW) return;
  int yy = p / cW, xx = p - yy * cW;
  float2 v = img[(yy + (cH - cW) / 2) * cW + xx];
  out[p * 2] = v.x; out[p * 2 + 1] = v.y;
}

// ---------------- host ----------------
extern "C" void kernel_launch(void* const* d_in, const int* in_sizes, int n_in,
                              void* d_out, int out_size, void* d_ws, size_t ws_size,
                              hipStream_t stream) {
  const float2* b    = (const float2*)d_in[0];
  const float*  smap = (const float*)d_in[1];
  const float*  w0   = (const float*)d_in[2];
  const float*  b0   = (const float*)d_in[3];
  const float*  w1   = (const float*)d_in[4];
  const float*  b1   = (const float*)d_in[5];
  const float*  g1   = (const float*)d_in[6];
  const float*  be1  = (const float*)d_in[7];
  const float*  w2   = (const float*)d_in[8];
  const float*  b2   = (const float*)d_in[9];
  const float*  g2   = (const float*)d_in[10];
  const float*  be2  = (const float*)d_in[11];
  const float*  wl   = (const float*)d_in[12];
  const float*  bl   = (const float*)d_in[13];
  const float*  sc   = (const float*)d_in[14];

  char* ws = (char*)d_ws;
  float*    SSm   = (float*)(ws + OFF_SS);
  float2*   Ahb   = (float2*)(ws + OFF_AHB);
  float2*   img   = (float2*)(ws + OFF_IMG);
  float2*   dv    = (float2*)(ws + OFF_D);
  float2*   qv    = (float2*)(ws + OFF_QD);
  float2*   Pm    = (float2*)(ws + OFF_P);
  float2*   Qm    = (float2*)(ws + OFF_Q);
  float*    small = (float*)(ws + OFF_SMALL);
  float* s1 = small + 0,  *t1 = small + 16;
  float* s2 = small + 32, *t2 = small + 48;
  float2* alpha = (float2*)(small + 64);
  float*  aPart = small + 128;
  _Float16* B0 = (_Float16*)(ws + OFF_WTS);
  _Float16* B1 = B0 + 32 * 16;
  _Float16* B2 = B1 + 160 * 16;
  _Float16* BL = B2 + 160 * 16;
  _Float16* y0 = (_Float16*)(ws + OFF_Y0);
  _Float16* y1 = (_Float16*)(ws + OFF_Y1);
  _Float16* y2 = (_Float16*)(ws + OFF_Y2);
  float2* ycoil = (float2*)(ws + OFF_YCOIL);
  float* bnSum = (float*)(ws + OFF_BNSUM);
  float* bnSq  = (float*)(ws + OFF_BNSQ);

  const int PW_BLK = (cH * cW + 255) / 256;   // 768

  k_prep_w<<<1, 256, 0, stream>>>(w0, w1, w2, wl, B0, B1, B2, BL);
  // ortho ifft2(b): columns (512-pt), then rows (384-pt)
  k_fft_col512<<<cC * cW, 256, 0, stream>>>(b, ycoil, +1.f);
  k_fft_row384<<<cC * cH, 192, 0, stream>>>(ycoil, +1.f);
  k_decode<<<PW_BLK, 256, 0, stream>>>(ycoil, smap, Ahb, img, SSm);

  for (int it = 0; it < 4; ++it) {
    k_conv0   <<<CONV_BLOCKS, 256, 0, stream>>>(img, B0, b0, y0);
    k_conv_mid<<<CONV_BLOCKS, 256, 0, stream>>>(y0, B1, b1, nullptr, nullptr, y1, bnSum, bnSq);
    k_bnp     <<<1, 16, 0, stream>>>(bnSum, bnSq, g1, be1, s1, t1);
    k_conv_mid<<<CONV_BLOCKS, 256, 0, stream>>>(y1, B2, b2, s1, t1, y2, bnSum, bnSq);
    k_bnp     <<<1, 16, 0, stream>>>(bnSum, bnSq, g2, be2, s2, t2);
    k_convL   <<<CONV_BLOCKS, 256, 0, stream>>>(y2, BL, s2, t2, bl, img);
    k_make_dqd<<<PW_BLK, 256, 0, stream>>>(img, Ahb, SSm, sc, it, dv, qv);
    k_gemm_pq <<<GEMM_BLOCKS, 256, 0, stream>>>(dv, qv, Pm, Qm);
    k_alpha   <<<ALPHA_BLOCKS, 256, 0, stream>>>(Pm, Qm, aPart);
    k_alpha_fin<<<1, 32, 0, stream>>>(aPart, alpha);
    k_update  <<<PW_BLK, 256, 0, stream>>>(img, dv, alpha);
  }
  k_crop<<<(cW * cW + 255) / 256, 256, 0, stream>>>(img, (float*)d_out);
}
